// MultiTaskModelMP_73358041415848
// MI455X (gfx1250) — matrix operand order repaired and checked by reading, not compiled, not run on hardware
//
#include <hip/hip_runtime.h>
#include <hip/hip_bf16.h>

// ---------------------------------------------------------------------------
// MI455X (gfx1250) multi-task MoE-style model, branch-gated.
// Dominant GEMM (131072x512 @ 512x512, branch-selected) via
// v_wmma_f32_16x16x32_bf16 with a bf16 hi/lo split (3 WMMAs / K-step) for
// ~f32 accuracy. Weights are pre-split once per launch into fragment-ordered
// bf16 hi/lo; activations are split once per workgroup into LDS fragment
// slots, so the hot loop is ds_load_b128 + global_load_b128 + WMMA only.
// ---------------------------------------------------------------------------

#define HIDDEN          512
#define N_NODES         131072
#define N_GRAPHS        4096
#define NODES_PER_GRAPH 32
#define GHD             50
#define NHD             3
#define N_BRANCH        4

// d_out layout (flat, f32): g_head | g_var | n_head | n_var
#define OFF_GHEAD 0
#define OFF_GVAR  (N_GRAPHS * GHD)                       // 204800
#define OFF_NHEAD (2 * N_GRAPHS * GHD)                   // 409600
#define OFF_NVAR  (2 * N_GRAPHS * GHD + N_NODES * NHD)   // 802816

// Fragment-ordered weight buffer: slot = (br*16 + kc)*32 + nt.
// Each slot = 1024 bf16 (hi 512 | lo 512); lane j's 16 bf16 at j*16.
#define SLOT_BF16   1024
#define BR_SLOTS    (16 * 32)

typedef __attribute__((ext_vector_type(16))) __bf16 v16bf;
typedef __attribute__((ext_vector_type(8)))  float  v8f;

union BF16x16 { v16bf v; unsigned short u[16]; uint4 q[2]; };

// Split 4 floats into packed bf16 hi/lo dword pairs (truncating split:
// f = hi + lo exactly at bf16-hi granularity; lo truncated -> ~2^-16 rel).
__device__ __forceinline__ void split_pack4(const float4 f, uint2& hp, uint2& lp) {
  unsigned int b0 = __float_as_uint(f.x), b1 = __float_as_uint(f.y);
  unsigned int b2 = __float_as_uint(f.z), b3 = __float_as_uint(f.w);
  unsigned int h0 = b0 & 0xffff0000u, h1 = b1 & 0xffff0000u;
  unsigned int h2 = b2 & 0xffff0000u, h3 = b3 & 0xffff0000u;
  unsigned int l0 = __float_as_uint(f.x - __uint_as_float(h0));
  unsigned int l1 = __float_as_uint(f.y - __uint_as_float(h1));
  unsigned int l2 = __float_as_uint(f.z - __uint_as_float(h2));
  unsigned int l3 = __float_as_uint(f.w - __uint_as_float(h3));
  hp.x = (h0 >> 16) | (h1 & 0xffff0000u);
  hp.y = (h2 >> 16) | (h3 & 0xffff0000u);
  lp.x = (l0 >> 16) | (l1 & 0xffff0000u);
  lp.y = (l2 >> 16) | (l3 & 0xffff0000u);
}

// WMMA A-layout mapping for element (m, k) inside a 16x32 tile:
// lane = (m&15) + ((k>>3)&1)*16 ; e = (k&7) + ((k>>4)&1)*8
// LDS A slot = 2048 B: hi 1024 | lo 1024 ; lane chunk = 32 B at lane*32.
__device__ __forceinline__ void lds_store_a_quad(char* smem, int slot, int m, int k,
                                                 uint2 hp, uint2 lp) {
  const int lane = (m & 15) + (((k >> 3) & 1) << 4);
  const int e    = (k & 7) + (((k >> 4) & 1) << 3);
  char* bp = smem + slot * 2048 + lane * 32 + e * 2;
  *(uint2*)(bp)        = hp;
  *(uint2*)(bp + 1024) = lp;
}

__device__ __forceinline__ void load_a_lds(const char* smem, int slot, int lane,
                                           v16bf& hi, v16bf& lo) {
  const char* p = smem + slot * 2048 + lane * 32;
  BF16x16 H, L;
  H.q[0] = *(const uint4*)(p);
  H.q[1] = *(const uint4*)(p + 16);
  L.q[0] = *(const uint4*)(p + 1024);
  L.q[1] = *(const uint4*)(p + 1040);
  hi = H.v; lo = L.v;
}

// B fragment from pre-split fragment-ordered buffer: 4x b128.
__device__ __forceinline__ void load_b_pre(const unsigned short* __restrict__ Wp,
                                           int slot, int lane, v16bf& hi, v16bf& lo) {
  const unsigned short* p = Wp + (size_t)slot * SLOT_BF16 + lane * 16;
  BF16x16 H, L;
  H.q[0] = *(const uint4*)(p);
  H.q[1] = *(const uint4*)(p + 8);
  L.q[0] = *(const uint4*)(p + 512);
  L.q[1] = *(const uint4*)(p + 520);
  hi = H.v; lo = L.v;
}

// bf16x3 product: C += Ahi*Bhi + Ahi*Blo + Alo*Bhi  (f32 accumulate)
__device__ __forceinline__ v8f wmma3(v16bf ah, v16bf al, v16bf bh, v16bf bl, v8f c) {
  c = __builtin_amdgcn_wmma_f32_16x16x32_bf16(false, ah, false, bh, (short)0, c, false, false);
  c = __builtin_amdgcn_wmma_f32_16x16x32_bf16(false, ah, false, bl, (short)0, c, false, false);
  c = __builtin_amdgcn_wmma_f32_16x16x32_bf16(false, al, false, bh, (short)0, c, false, false);
  return c;
}

__device__ __forceinline__ void zero8(v8f& v) {
#pragma unroll
  for (int i = 0; i < 8; ++i) v[i] = 0.0f;
}

// ---------------------------------------------------------------------------
// Kernel 0: pre-split a (N_BRANCH x 512 x 512) f32 weight matrix into
// fragment-ordered bf16 hi/lo (B layout: lane = n&15 + ((k>>3)&1)*16, e=(k&7)+((k>>4)&1)*8).
// ---------------------------------------------------------------------------
__global__ __launch_bounds__(256) void prep_weights_kernel(const float* __restrict__ W,
                                                           unsigned int* __restrict__ Wp) {
  const int id = blockIdx.x * 256 + threadIdx.x;   // 2^19 total
  const int n  = id & 511;
  const int kp = (id >> 9) & 255;
  const int br = id >> 17;
  const int k0 = kp * 2;

  const float f0 = W[((size_t)br * HIDDEN + k0) * HIDDEN + n];
  const float f1 = W[((size_t)br * HIDDEN + k0 + 1) * HIDDEN + n];
  unsigned int b0 = __float_as_uint(f0);
  unsigned int b1 = __float_as_uint(f1);
  unsigned int h0 = b0 & 0xffff0000u, h1 = b1 & 0xffff0000u;
  unsigned int l0 = __float_as_uint(f0 - __uint_as_float(h0));
  unsigned int l1 = __float_as_uint(f1 - __uint_as_float(h1));
  unsigned int hi = (h0 >> 16) | (h1 & 0xffff0000u);
  unsigned int lo = (l0 >> 16) | (l1 & 0xffff0000u);

  const int kc   = k0 >> 5;
  const int kw   = k0 & 31;
  const int lane = (n & 15) + (((kw >> 3) & 1) << 4);
  const int e    = (kw & 7) + (((kw >> 4) & 1) << 3); // even
  const int nt   = n >> 4;
  const size_t slot = (size_t)((br * 16 + kc) * 32 + nt);
  unsigned int* dst = Wp + slot * (SLOT_BF16 / 2) + lane * 8 + (e >> 1);
  dst[0]   = hi;                                   // hi half
  dst[256] = lo;                                   // lo half (+512 bf16)
}

// ---------------------------------------------------------------------------
// Kernel 1: per-graph mean pooling  x (131072x512) -> x_graph (4096x512)
// ---------------------------------------------------------------------------
__global__ __launch_bounds__(256) void pool_kernel(const float* __restrict__ x,
                                                   float* __restrict__ xg) {
  const int g = blockIdx.x;
  const int c = threadIdx.x * 2;
  const float* p = x + (size_t)g * NODES_PER_GRAPH * HIDDEN + c;
  float sx = 0.f, sy = 0.f;
#pragma unroll 8
  for (int r = 0; r < NODES_PER_GRAPH; ++r) {
    float2 v = *(const float2*)(p + (size_t)r * HIDDEN);
    sx += v.x; sy += v.y;
  }
  const float inv = 1.0f / (float)NODES_PER_GRAPH;
  xg[(size_t)g * HIDDEN + c]     = sx * inv;
  xg[(size_t)g * HIDDEN + c + 1] = sy * inv;
}

// ---------------------------------------------------------------------------
// Kernel 2 (node path): one WG per graph (32 nodes share one branch).
// H = relu(X(32x512) @ Wn1[b] + bn1[b])  via WMMA bf16x3, fused with
// on = H @ Wn2[b] + bn2[b]  ->  n_head / n_var written directly.
// LDS: [0,64K) A-fragments during GEMM; re-used as Hs|W2s in the epilogue.
// ---------------------------------------------------------------------------
__global__ __launch_bounds__(256) void node_kernel(
    const float* __restrict__ x,
    const unsigned short* __restrict__ Wn1p, const float* __restrict__ bn1,
    const float* __restrict__ Wn2, const float* __restrict__ bn2,
    const int* __restrict__ ds, float* __restrict__ out) {
  __shared__ __attribute__((aligned(16))) char smem[65536];

  const int g    = blockIdx.x;
  const int t    = threadIdx.x;
  const int w    = t >> 5;              // wave id (8 waves)
  const int lane = t & 31;
  const int br   = ds[g];
  const unsigned short* Wp = Wn1p + (size_t)br * BR_SLOTS * SLOT_BF16;
  const int row0 = g * NODES_PER_GRAPH;

  // Cooperative A prep: 32 rows x 512 cols -> 32 frag slots (kc*2 + mt).
  // Coalesced reads: quad id runs contiguously along rows.
#pragma unroll
  for (int q = 0; q < 16; ++q) {
    const int qid = t + 256 * q;        // 0..4095
    const int m   = qid >> 7;           // 0..31
    const int k   = (qid & 127) << 2;   // 0..508, step 4
    float4 f4 = *(const float4*)(x + (size_t)(row0 + m) * HIDDEN + k);
    uint2 hp, lp;
    split_pack4(f4, hp, lp);
    lds_store_a_quad(smem, (k >> 5) * 2 + (m >> 4), m, k & 31, hp, lp);
  }
  __syncthreads();

  v8f acc[2][4];
#pragma unroll
  for (int mt = 0; mt < 2; ++mt)
#pragma unroll
    for (int nt = 0; nt < 4; ++nt) zero8(acc[mt][nt]);

  // Main K loop: 16 chunks of K=32. Each wave: 2 M-tiles x 4 N-tiles.
#pragma unroll 2
  for (int kc = 0; kc < HIDDEN / 32; ++kc) {
    v16bf a0h, a0l, a1h, a1l;
    load_a_lds(smem, kc * 2 + 0, lane, a0h, a0l);
    load_a_lds(smem, kc * 2 + 1, lane, a1h, a1l);
#pragma unroll
    for (int nt = 0; nt < 4; ++nt) {
      v16bf bh, bl;
      load_b_pre(Wp, kc * 32 + (w * 4 + nt), lane, bh, bl);
      acc[0][nt] = wmma3(a0h, a0l, bh, bl, acc[0][nt]);
      acc[1][nt] = wmma3(a1h, a1l, bh, bl, acc[1][nt]);
    }
  }
  __syncthreads();                      // A-frag reads done; smem re-used below

  float* Hs  = (float*)smem;            // 16 x 516 f32 = 33024 B
  float* W2s = (float*)(smem + 33024);  // 512 x 6 f32  = 12288 B

  for (int idx = t; idx < HIDDEN * 6; idx += 256)
    W2s[idx] = Wn2[(size_t)br * HIDDEN * 6 + idx];

  const int half = lane >> 4;
  const int nl   = lane & 15;
  float bias[4];
#pragma unroll
  for (int nt = 0; nt < 4; ++nt)
    bias[nt] = bn1[br * HIDDEN + (w * 4 + nt) * 16 + nl];

  // Two 16-row phases: bias+relu -> Hs, then 512->6 dot + head/var writes.
  for (int mt = 0; mt < 2; ++mt) {
    if (mt) __syncthreads();            // previous phase's reads done
#pragma unroll
    for (int nt = 0; nt < 4; ++nt) {
      const int nAbs = (w * 4 + nt) * 16 + nl;
#pragma unroll
      for (int v = 0; v < 8; ++v) {
        // C/D layout: lane = N column, VGPR v = row (half*8 + v)
        float h = acc[mt][nt][v] + bias[nt];
        Hs[(half * 8 + v) * 516 + nAbs] = h > 0.f ? h : 0.f;
      }
    }
    __syncthreads();
    if (t < 96) {                       // 16 rows x 6 outputs
      const int m = t & 15, o = t >> 4;
      float a2 = bn2[br * 6 + o];
      const float* hr = Hs + m * 516;
#pragma unroll 8
      for (int n = 0; n < HIDDEN; ++n) a2 += hr[n] * W2s[n * 6 + o];
      const size_t row = (size_t)(row0 + mt * 16 + m);
      if (o < NHD) out[OFF_NHEAD + row * NHD + o]         = a2;
      else         out[OFF_NVAR  + row * NHD + (o - NHD)] = a2 * a2;
    }
  }
}

// ---------------------------------------------------------------------------
// Kernel 3 (graph path, layer 1): WG per (16-graph tile, branch).
// H = relu(x_graph @ Ws[b] + bs[b]); rows stored to Hg only where
// dataset_ids[g]==b (disjoint writers -> no atomics).
// ---------------------------------------------------------------------------
__global__ __launch_bounds__(256) void graph_mm1_kernel(
    const float* __restrict__ xg,
    const unsigned short* __restrict__ Wsp, const float* __restrict__ bs,
    const int* __restrict__ ds, float* __restrict__ Hg) {
  __shared__ __attribute__((aligned(16))) char smemA[32768];  // 16 A-frag slots
  __shared__ int dss[16];
  const int tile = blockIdx.x;          // 0..255
  const int br   = blockIdx.y;          // 0..3
  const int t    = threadIdx.x;
  const int row0 = tile * 16;

  if (t < 16) dss[t] = ds[row0 + t];
  __syncthreads();
  bool any = false;
#pragma unroll
  for (int i = 0; i < 16; ++i) any |= (dss[i] == br);
  if (!any) return;                     // block-uniform: EXEC stays all-ones

  const int w    = t >> 5;
  const int lane = t & 31;
  const unsigned short* Wp = Wsp + (size_t)br * BR_SLOTS * SLOT_BF16;

  // Cooperative A prep: 16 rows x 512 cols -> 16 frag slots (kc).
#pragma unroll
  for (int q = 0; q < 8; ++q) {
    const int qid = t + 256 * q;        // 0..2047
    const int m   = qid >> 7;           // 0..15
    const int k   = (qid & 127) << 2;
    float4 f4 = *(const float4*)(xg + (size_t)(row0 + m) * HIDDEN + k);
    uint2 hp, lp;
    split_pack4(f4, hp, lp);
    lds_store_a_quad(smemA, k >> 5, m, k & 31, hp, lp);
  }
  __syncthreads();

  v8f acc[4];
#pragma unroll
  for (int nt = 0; nt < 4; ++nt) zero8(acc[nt]);

#pragma unroll 2
  for (int kc = 0; kc < HIDDEN / 32; ++kc) {
    v16bf ah, al;
    load_a_lds(smemA, kc, lane, ah, al);
#pragma unroll
    for (int nt = 0; nt < 4; ++nt) {
      v16bf bh, bl;
      load_b_pre(Wp, kc * 32 + (w * 4 + nt), lane, bh, bl);
      acc[nt] = wmma3(ah, al, bh, bl, acc[nt]);
    }
  }

  const int half = lane >> 4;
  const int nl   = lane & 15;
#pragma unroll
  for (int nt = 0; nt < 4; ++nt) {
    const int nAbs = (w * 4 + nt) * 16 + nl;
    const float bias = bs[br * HIDDEN + nAbs];
#pragma unroll
    for (int v = 0; v < 8; ++v) {
      const int m = half * 8 + v;
      if (dss[m] == br) {
        float h = acc[nt][v] + bias;
        Hg[(size_t)(row0 + m) * HIDDEN + nAbs] = h > 0.f ? h : 0.f;
      }
    }
  }
}

// ---------------------------------------------------------------------------
// Kernel 4 (graph path, layer 2): og = Hg[g] @ Wg[b] + bg[b]; head/var^2.
// ---------------------------------------------------------------------------
__global__ __launch_bounds__(128) void graph_head_kernel(
    const float* __restrict__ Hg,
    const float* __restrict__ Wg, const float* __restrict__ bg,
    const int* __restrict__ ds, float* __restrict__ out) {
  const int g = blockIdx.x;
  const int o = threadIdx.x;
  if (o >= 2 * GHD) return;
  const int br = ds[g];
  const float* wp = Wg + (size_t)br * HIDDEN * (2 * GHD) + o;
  const float* h  = Hg + (size_t)g * HIDDEN;
  float acc = bg[br * (2 * GHD) + o];
#pragma unroll 8
  for (int n = 0; n < HIDDEN; ++n) acc += h[n] * wp[(size_t)n * (2 * GHD)];
  if (o < GHD) out[OFF_GHEAD + (size_t)g * GHD + o]         = acc;
  else         out[OFF_GVAR  + (size_t)g * GHD + (o - GHD)] = acc * acc;
}

// ---------------------------------------------------------------------------
extern "C" void kernel_launch(void* const* d_in, const int* in_sizes, int n_in,
                              void* d_out, int out_size, void* d_ws, size_t ws_size,
                              hipStream_t stream) {
  const float* x   = (const float*)d_in[0];
  const float* Ws  = (const float*)d_in[1];
  const float* bs  = (const float*)d_in[2];
  const float* Wg  = (const float*)d_in[3];
  const float* bg  = (const float*)d_in[4];
  const float* Wn1 = (const float*)d_in[5];
  const float* bn1 = (const float*)d_in[6];
  const float* Wn2 = (const float*)d_in[7];
  const float* bn2 = (const float*)d_in[8];
  const int*   ds  = (const int*)d_in[10];
  float* out = (float*)d_out;

  // scratch: x_graph (8MB) | Hg (8MB) | Wn1 frags (4MB) | Ws frags (4MB)
  char* wsb = (char*)d_ws;
  float* xg = (float*)wsb;
  float* Hg = (float*)(wsb + (size_t)N_GRAPHS * HIDDEN * 4);
  unsigned int* Wn1p = (unsigned int*)(wsb + (size_t)2 * N_GRAPHS * HIDDEN * 4);
  unsigned int* Wsp  = (unsigned int*)(wsb + (size_t)2 * N_GRAPHS * HIDDEN * 4 +
                                       (size_t)N_BRANCH * BR_SLOTS * SLOT_BF16 * 2);

  prep_weights_kernel<<<2048, 256, 0, stream>>>(Wn1, Wn1p);
  prep_weights_kernel<<<2048, 256, 0, stream>>>(Ws,  Wsp);
  pool_kernel        <<<N_GRAPHS, 256, 0, stream>>>(x, xg);
  graph_mm1_kernel   <<<dim3(N_GRAPHS / 16, 4), 256, 0, stream>>>(
                        xg, (const unsigned short*)Wsp, bs, ds, Hg);
  graph_head_kernel  <<<N_GRAPHS, 128, 0, stream>>>(Hg, Wg, bg, ds, out);
  node_kernel        <<<N_GRAPHS, 256, 0, stream>>>(
                        x, (const unsigned short*)Wn1p, bn1, Wn2, bn2, ds, out);
}
